// HAT_13657996002163
// MI455X (gfx1250) — matrix operand order repaired and checked by reading, not compiled, not run on hardware
//
#include <hip/hip_runtime.h>
#include <hip/hip_bf16.h>

// ---------------- problem constants (match reference) ----------------
constexpr int   kN      = 50000;     // nodes
constexpr int   kE      = 1600000;   // edges per metapath
constexpr int   kNFEAT  = 256;
constexpr int   kNHID   = 64;
constexpr int   kNHEADS = 8;
constexpr int   kNSEM   = 2;
constexpr int   kNMP    = 128;
constexpr int   kNLABEL = 8;
constexpr float kALPHA  = 0.2f;
constexpr int   kD      = kNHID * kNHEADS;   // 512 per metapath
constexpr int   kZW     = kNSEM * kD;        // 1024 z row width

typedef __attribute__((ext_vector_type(16))) _Float16 v16h;
typedef __attribute__((ext_vector_type(8)))  _Float16 v8h;
typedef __attribute__((ext_vector_type(8)))  float    v8f;
typedef __attribute__((ext_vector_type(4)))  int      v4i;
typedef __attribute__((address_space(1))) v4i* gv4i_p;   // global <4 x i32>*
typedef __attribute__((address_space(3))) v4i* lv4i_p;   // LDS    <4 x i32>*

#define AF_CAT 0, 1, 2, 3, 4, 5, 6, 7, 8, 9, 10, 11, 12, 13, 14, 15

// ---------------------------------------------------------------------
// Elementwise f32 -> f16 convert (one-off; makes GEMM A side pure b128)
// ---------------------------------------------------------------------
__global__ __launch_bounds__(256)
void han_cvt_f16_kernel(const float* __restrict__ src, _Float16* __restrict__ dst,
                        long long count) {
    long long i = (long long)blockIdx.x * 256 + threadIdx.x;
    if (i < count) dst[i] = (_Float16)src[i];
}

// ---------------------------------------------------------------------
// Pack B matrix [K x Ncols] f32 into WMMA B-fragment order (f16):
// fragment (kt,cti): lane l holds col cti*16+(l&15),
//   half h -> K = kt*32 + ((l>=16)?8:0) + (h&7) + ((h>=8)?16:0)   (per ISA 16-bit B 32x16 layout)
// mode 0: B is W viewed as [256, 512] for one metapath
//         col = j*64+c  ->  W[((sem*8+j)*256+k)*64+c]
// mode 1: plain row-major [K x Ncols]
// ---------------------------------------------------------------------
__global__ __launch_bounds__(256)
void han_pack_b_kernel(const float* __restrict__ B, _Float16* __restrict__ Bp,
                       int K, int Ncols, int mode, int sem) {
    int t = blockIdx.x * 256 + threadIdx.x;
    int total = K * Ncols;
    if (t >= total) return;
    int frag   = t >> 9;        // /512 elements per fragment
    int within = t & 511;
    int lane   = within >> 4;
    int hh     = within & 15;
    int NT     = Ncols >> 4;
    int kt     = frag / NT;
    int cti    = frag - kt * NT;
    int k      = kt * 32 + ((lane >> 4) << 3) + (hh & 7) + ((hh >> 3) << 4);
    int n      = cti * 16 + (lane & 15);
    float v;
    if (mode == 0) {
        int j = n >> 6, c = n & 63;
        v = B[(((size_t)sem * kNHEADS + j) * kNFEAT + k) * kNHID + c];
    } else {
        v = B[(size_t)k * Ncols + n];
    }
    Bp[t] = (_Float16)v;
}

// ---------------------------------------------------------------------
// WMMA GEMM: C[M x Ncols] = A[M x K] (f16 row-major) * Bpack (f16 frags)
// Block = 256 threads = 8 waves; block tile 256(M) x 64(N).
// Per wave: two 16-row tiles x four 16-col tiles = 8 accumulators;
// 8 WMMAs per K-step against 8 ds_load_b128 (B panel staged once in LDS
// via CDNA5 async global->LDS copies, ASYNCcnt) + 4 global b128 A loads.
// ---------------------------------------------------------------------
__global__ __launch_bounds__(256)
void han_wmma_gemm_kernel(const _Float16* __restrict__ A,
                          const _Float16* __restrict__ Bp,
                          float* __restrict__ C, int M, int K, int Ncols) {
    extern __shared__ _Float16 ldsB[];      // KT * 2048 halfs (KT * 4KB)
    const int lane = threadIdx.x & 31;
    const int wave = threadIdx.x >> 5;
    const int NT   = Ncols >> 4;
    const int KT   = K >> 5;
    const int m0   = (blockIdx.x * 8 + wave) * 32;
    const int n0   = blockIdx.y * 64;

    // ---- stage B panel: per kt, 4 contiguous fragments = 4KB ----
    for (int kt = 0; kt < KT; ++kt) {
        const _Float16* gsrc = Bp + ((size_t)kt * NT + (n0 >> 4)) * 512;
        _Float16*       ldst = ldsB + kt * 2048;
        const int tid = threadIdx.x;        // 256 thr x 16B = 4KB
#if __has_builtin(__builtin_amdgcn_global_load_async_to_lds_b128)
        __builtin_amdgcn_global_load_async_to_lds_b128(
            (gv4i_p)(gsrc + tid * 8),
            (lv4i_p)(ldst + tid * 8),
            0, 0);
#else
        *(float4*)(ldst + tid * 8) = *(const float4*)(gsrc + tid * 8);
#endif
    }
#if __has_builtin(__builtin_amdgcn_s_wait_asynccnt)
    __builtin_amdgcn_s_wait_asynccnt(0);
#endif
    __syncthreads();

    if (m0 >= M) return;                    // wave-uniform exit (EXEC stays full)

    // A 16-bit 16x32 fragment layout: lane l -> row (l&15), kb = (l>=16)?8:0;
    // halves 0..7 at K=kb+0..7, halves 8..15 at K=16+kb+0..7 (two 16B runs).
    const int kb    = (lane >> 4) << 3;
    const int rowA0 = m0 + (lane & 15);
    const int rowA1x = m0 + 16 + (lane & 15);
    const int rowA1 = rowA1x < M ? rowA1x : rowA0;   // clamp for M tail
    const bool tile1 = (m0 + 16) < M;
    const _Float16* a0 = A + (size_t)rowA0 * K + kb;
    const _Float16* a1 = A + (size_t)rowA1 * K + kb;

    v8f acc00 = {}, acc01 = {}, acc02 = {}, acc03 = {};
    v8f acc10 = {}, acc11 = {}, acc12 = {}, acc13 = {};

    v8h alo0 = *(const v8h*)(a0);
    v8h ahi0 = *(const v8h*)(a0 + 16);
    v8h alo1 = *(const v8h*)(a1);
    v8h ahi1 = *(const v8h*)(a1 + 16);

    for (int kt = 0; kt < KT; ++kt) {
        v16h af0 = __builtin_shufflevector(alo0, ahi0, AF_CAT);
        v16h af1 = __builtin_shufflevector(alo1, ahi1, AF_CAT);

        if (kt + 1 < KT) {                  // software-pipeline next A tiles
            const int o = (kt + 1) * 32;
            __builtin_prefetch((const void*)(a0 + o + 32), 0, 1);
            alo0 = *(const v8h*)(a0 + o);
            ahi0 = *(const v8h*)(a0 + o + 16);
            alo1 = *(const v8h*)(a1 + o);
            ahi1 = *(const v8h*)(a1 + o + 16);
        }

        const _Float16* lb = ldsB + kt * 2048 + lane * 16;
        v16h b0 = *(const v16h*)(lb);
        v16h b1 = *(const v16h*)(lb + 512);
        v16h b2 = *(const v16h*)(lb + 1024);
        v16h b3 = *(const v16h*)(lb + 1536);

        // same A operand 4x back-to-back -> reuse_a hint on trailing ops
        acc00 = __builtin_amdgcn_wmma_f32_16x16x32_f16(false, af0, false, b0, (short)0, acc00, false, false);
        acc01 = __builtin_amdgcn_wmma_f32_16x16x32_f16(false, af0, false, b1, (short)0, acc01, true,  false);
        acc02 = __builtin_amdgcn_wmma_f32_16x16x32_f16(false, af0, false, b2, (short)0, acc02, true,  false);
        acc03 = __builtin_amdgcn_wmma_f32_16x16x32_f16(false, af0, false, b3, (short)0, acc03, true,  false);
        acc10 = __builtin_amdgcn_wmma_f32_16x16x32_f16(false, af1, false, b0, (short)0, acc10, false, false);
        acc11 = __builtin_amdgcn_wmma_f32_16x16x32_f16(false, af1, false, b1, (short)0, acc11, true,  false);
        acc12 = __builtin_amdgcn_wmma_f32_16x16x32_f16(false, af1, false, b2, (short)0, acc12, true,  false);
        acc13 = __builtin_amdgcn_wmma_f32_16x16x32_f16(false, af1, false, b3, (short)0, acc13, true,  false);
    }

    // C/D layout: lane l -> col (l&15); VGPR r -> row base + ((l>=16)?8:0) + r
    const int colb = n0 + (lane & 15);
    const int rowb0 = m0 + ((lane >> 4) << 3);
#pragma unroll
    for (int r = 0; r < 8; ++r) {
        float* crow = C + (size_t)(rowb0 + r) * Ncols + colb;
        crow[0]  = acc00[r];
        crow[16] = acc01[r];
        crow[32] = acc02[r];
        crow[48] = acc03[r];
    }
    if (tile1) {
        const int rowb1 = rowb0 + 16;
#pragma unroll
        for (int r = 0; r < 8; ++r) {
            float* crow = C + (size_t)(rowb1 + r) * Ncols + colb;
            crow[0]  = acc10[r];
            crow[16] = acc11[r];
            crow[32] = acc12[r];
            crow[48] = acc13[r];
        }
    }
}

// ---------------------------------------------------------------------
// Attention score dots: s_src[h,n] = h_row . a[:64], s_dst[h,n] = h_row . a[64:]
// One wave per (node, head); lane covers 2 channels; shfl reduction.
// ---------------------------------------------------------------------
__global__ __launch_bounds__(256)
void han_score_kernel(const float* __restrict__ h, const float* __restrict__ a,
                      float* __restrict__ s_src, float* __restrict__ s_dst, int sem) {
    const int wave = threadIdx.x >> 5, lane = threadIdx.x & 31;
    long long wg = (long long)blockIdx.x * 8 + wave;
    if (wg >= (long long)kN * kNHEADS) return;
    int n = (int)(wg >> 3), head = (int)(wg & 7);
    const float* hr = h + (size_t)n * kD + head * kNHID;
    const float* av = a + ((size_t)sem * kNHEADS + head) * (2 * kNHID);
    float ss = hr[lane] * av[lane]      + hr[lane + 32] * av[lane + 32];
    float sd = hr[lane] * av[64 + lane] + hr[lane + 32] * av[96 + lane];
#pragma unroll
    for (int off = 16; off; off >>= 1) {
        ss += __shfl_down(ss, off, 32);
        sd += __shfl_down(sd, off, 32);
    }
    if (lane == 0) { s_src[head * kN + n] = ss; s_dst[head * kN + n] = sd; }
}

// ---------------------------------------------------------------------
// Edge scatter: one wave per (edge, head). e = exp(-leakyrelu(logit)).
// denom[src] += e; num[src, sem*512+head*64+c] += e * h[dst, head*64+c]
// lane = channel -> coalesced gather of h[dst] and coalesced atomics.
// This is the roofline term: ~26 GB of gather+RMW; h stays hot in L2.
// ---------------------------------------------------------------------
__global__ __launch_bounds__(256)
void han_edge_kernel(const long long* __restrict__ src, const long long* __restrict__ dst,
                     const float* __restrict__ s_src, const float* __restrict__ s_dst,
                     const float* __restrict__ h, float* __restrict__ z,
                     float* __restrict__ denom, int sem) {
    const int wave = threadIdx.x >> 5, lane = threadIdx.x & 31;
    long long wg = (long long)blockIdx.x * 8 + wave;   // over E * NHEADS
    if (wg >= (long long)kE * kNHEADS) return;
    long long e = wg >> 3;
    int head = (int)(wg & 7);
    int sn = (int)src[e], dn = (int)dst[e];
    float logit = s_src[head * kN + sn] + s_dst[head * kN + dn];
    float lr = logit > 0.f ? logit : kALPHA * logit;
    float ev = __expf(-lr);
    if (lane == 0) atomicAdd(&denom[(sem * kNHEADS + head) * kN + sn], ev);
    const float* hd = h + (size_t)dn * kD + head * kNHID;
    float* np = z + (size_t)sn * kZW + sem * kD + head * kNHID;
    atomicAdd(&np[lane],      ev * hd[lane]);
    atomicAdd(&np[lane + 32], ev * hd[lane + 32]);
}

// ---------------------------------------------------------------------
// z = elu(num / (denom + 1e-16)) in place over [N, 2, 512];
// also emits zh = (f16)z laid out as [2N, 512] rows (n*2+p) for GEMM2 A.
// ---------------------------------------------------------------------
__global__ __launch_bounds__(256)
void han_z_finalize_kernel(float* __restrict__ z, _Float16* __restrict__ zh,
                           const float* __restrict__ denom) {
    long long idx = (long long)blockIdx.x * 256 + threadIdx.x;
    if (idx >= (long long)kN * kZW) return;
    int n = (int)(idx >> 10);
    int rem = (int)(idx & 1023);
    int sem = rem >> 9;
    int head = (rem >> 6) & 7;
    int col = rem & 511;
    float d = denom[(sem * kNHEADS + head) * kN + n] + 1e-16f;
    float v = z[idx] / d;
    v = v > 0.f ? v : (__expf(v) - 1.f);
    z[idx] = v;
    zh[((size_t)n * 2 + sem) * kD + col] = (_Float16)v;
}

// ---------------------------------------------------------------------
// Semantic score: row r of t[2N,128]: w = tanh(t+bp) . q; wsum[p] += w
// ---------------------------------------------------------------------
__global__ __launch_bounds__(256)
void han_sem_score_kernel(const float* __restrict__ t, const float* __restrict__ bp,
                          const float* __restrict__ q, float* __restrict__ wsum) {
    const int wave = threadIdx.x >> 5, lane = threadIdx.x & 31;
    long long r = (long long)blockIdx.x * 8 + wave;
    if (r >= 2LL * kN) return;
    const float* tr = t + r * kNMP;
    float acc = 0.f;
#pragma unroll
    for (int i = 0; i < 4; ++i) {
        int j = lane + i * 32;
        acc += tanhf(tr[j] + bp[j]) * q[j];
    }
#pragma unroll
    for (int off = 16; off; off >>= 1) acc += __shfl_down(acc, off, 32);
    if (lane == 0) atomicAdd(&wsum[(int)(r & 1)], acc);
}

__global__ void han_beta_kernel(const float* __restrict__ wsum, float* __restrict__ beta) {
    float m0 = wsum[0] / (float)kN, m1 = wsum[1] / (float)kN;
    float mx = fmaxf(m0, m1);
    float e0 = __expf(m0 - mx), e1 = __expf(m1 - mx);
    float inv = 1.f / (e0 + e1);
    beta[0] = e0 * inv;
    beta[1] = e1 * inv;
}

// ---------------------------------------------------------------------
// out[n,:] = sigmoid((beta0*z[n,0,:] + beta1*z[n,1,:]) @ Wc)   (one wave/node)
// ---------------------------------------------------------------------
__global__ __launch_bounds__(256)
void han_output_kernel(const float* __restrict__ z, const float* __restrict__ beta,
                       const float* __restrict__ Wc, float* __restrict__ out) {
    const int wave = threadIdx.x >> 5, lane = threadIdx.x & 31;
    long long n = (long long)blockIdx.x * 8 + wave;
    if (n >= kN) return;
    float b0 = beta[0], b1 = beta[1];
    const float* z0 = z + n * kZW;
    float acc[kNLABEL] = {};
#pragma unroll 4
    for (int i = 0; i < 16; ++i) {
        int c = lane * 16 + i;               // lane-owned slice of 512 channels
        float v = b0 * z0[c] + b1 * z0[kD + c];
        const float* wr = Wc + (size_t)c * kNLABEL;
#pragma unroll
        for (int d = 0; d < kNLABEL; ++d) acc[d] += v * wr[d];
    }
#pragma unroll
    for (int d = 0; d < kNLABEL; ++d)
#pragma unroll
        for (int off = 16; off; off >>= 1) acc[d] += __shfl_down(acc[d], off, 32);
    if (lane == 0) {
#pragma unroll
        for (int d = 0; d < kNLABEL; ++d)
            out[n * kNLABEL + d] = 1.f / (1.f + __expf(-acc[d]));
    }
}

// ---------------------------------------------------------------------
extern "C" void kernel_launch(void* const* d_in, const int* in_sizes, int n_in,
                              void* d_out, int out_size, void* d_ws, size_t ws_size,
                              hipStream_t stream) {
    const float*     x    = (const float*)d_in[0];
    const long long* adjs = (const long long*)d_in[1];   // int64 [NSEM,2,E]
    const float*     W    = (const float*)d_in[2];
    const float*     a    = (const float*)d_in[3];
    const float*     Wp   = (const float*)d_in[4];
    const float*     bpv  = (const float*)d_in[5];
    const float*     q    = (const float*)d_in[6];
    const float*     Wc   = (const float*)d_in[7];
    float* out = (float*)d_out;

    // ---- workspace carve-up (~445 MB) ----
    char* ws = (char*)d_ws;
    size_t off = 0;
    auto carve = [&](size_t bytes) -> char* {
        char* p = ws + off;
        off += (bytes + 255) & ~(size_t)255;
        return p;
    };
    float*    z     = (float*)carve((size_t)kN * kZW * 4);        // 204.8 MB (num -> z)
    float*    hbuf  = (float*)carve((size_t)kN * kD * 4);         // 102.4 MB (h, later t)
    _Float16* zh    = (_Float16*)carve((size_t)2 * kN * kD * 2);  // 102.4 MB f16 z
    _Float16* xh    = (_Float16*)carve((size_t)kN * kNFEAT * 2);  //  25.6 MB f16 x
    float*    denom = (float*)carve((size_t)kNSEM * kNHEADS * kN * 4);
    float*    ssrc  = (float*)carve((size_t)kNHEADS * kN * 4);
    float*    sdst  = (float*)carve((size_t)kNHEADS * kN * 4);
    _Float16* bpk1  = (_Float16*)carve((size_t)kNFEAT * kD * 2);  // packed W  (per sem)
    _Float16* bpk2  = (_Float16*)carve((size_t)kD * kNMP * 2);    // packed Wp
    float*    wsum  = (float*)carve(256);
    float*    beta  = (float*)carve(256);
    float*    tbuf  = hbuf;  // t[2N,128] aliases dead h buffer

    // ---- zero accumulators (memset nodes are graph-capturable) ----
    (void)hipMemsetAsync(z,     0, (size_t)kN * kZW * 4, stream);
    (void)hipMemsetAsync(denom, 0, (size_t)kNSEM * kNHEADS * kN * 4, stream);
    (void)hipMemsetAsync(wsum,  0, 8, stream);

    // ---- one-off: x -> f16 (A side of GEMM1, reused by both metapaths) ----
    {
        long long cnt = (long long)kN * kNFEAT;
        han_cvt_f16_kernel<<<(int)((cnt + 255) / 256), 256, 0, stream>>>(x, xh, cnt);
    }

    // ---- per-metapath GAT: keep h hot in L2 while its edges stream ----
    for (int sem = 0; sem < kNSEM; ++sem) {
        han_pack_b_kernel<<<(kNFEAT * kD + 255) / 256, 256, 0, stream>>>(
            W, bpk1, kNFEAT, kD, 0, sem);
        han_wmma_gemm_kernel<<<dim3((kN + 255) / 256, kD / 64), 256,
                               (kNFEAT / 32) * 4096, stream>>>(
            xh, bpk1, hbuf, kN, kNFEAT, kD);
        han_score_kernel<<<(kN * kNHEADS + 7) / 8, 256, 0, stream>>>(
            hbuf, a, ssrc, sdst, sem);
        const long long* srcp = adjs + (size_t)sem * 2 * kE;
        const long long* dstp = srcp + kE;
        han_edge_kernel<<<(int)(((long long)kE * kNHEADS + 7) / 8), 256, 0, stream>>>(
            srcp, dstp, ssrc, sdst, hbuf, z, denom, sem);
    }

    // ---- elu-normalize z (emit f16 copy), then HAN semantic attention ----
    han_z_finalize_kernel<<<(int)(((long long)kN * kZW + 255) / 256), 256, 0, stream>>>(
        z, zh, denom);

    han_pack_b_kernel<<<(kD * kNMP + 255) / 256, 256, 0, stream>>>(Wp, bpk2, kD, kNMP, 1, 0);
    han_wmma_gemm_kernel<<<dim3((2 * kN + 255) / 256, kNMP / 64), 256,
                           (kD / 32) * 4096, stream>>>(
        zh, bpk2, tbuf, 2 * kN, kD, kNMP);
    han_sem_score_kernel<<<(2 * kN + 7) / 8, 256, 0, stream>>>(tbuf, bpv, q, wsum);
    han_beta_kernel<<<1, 1, 0, stream>>>(wsum, beta);

    // ---- beta-mix + classifier + sigmoid ----
    han_output_kernel<<<(kN + 7) / 8, 256, 0, stream>>>(z, beta, Wc, out);

    (void)in_sizes; (void)n_in; (void)out_size; (void)ws_size;
}
